// CausalSelfAttention_62277025792433
// MI455X (gfx1250) — matrix operand order repaired and checked by reading, not compiled, not run on hardware
//
#include <hip/hip_runtime.h>
#include <math.h>

// ---- problem constants (match reference) ----
#define D_MODEL 1024
#define N_HEADS 16
#define HEAD_DIM 64
#define BATCH 2
#define SEQ 2048
#define M_TOTAL (BATCH * SEQ)   // 4096
#define N_QKV (3 * D_MODEL)     // 3072

typedef __attribute__((ext_vector_type(16))) __bf16 v16bf;
typedef __attribute__((ext_vector_type(8)))  float  v8f;

// ---- TDM availability (arity differs across toolchains) ----
#if defined(__has_builtin)
#if __has_builtin(__builtin_amdgcn_tensor_load_to_lds) && \
    __has_builtin(__builtin_amdgcn_s_wait_tensorcnt)
#define HAVE_TDM 1
#endif
#endif
#ifndef HAVE_TDM
#define HAVE_TDM 0
#endif

typedef unsigned int u32x4 __attribute__((ext_vector_type(4)));
typedef int          i32x4 __attribute__((ext_vector_type(4)));
typedef int          i32x8 __attribute__((ext_vector_type(8)));

union Frag {
  v16bf v;
  uint4 q[2];
  unsigned int   u32[8];
  unsigned short u16[16];
};

static __device__ __forceinline__ unsigned short f2bf(float f) {
  unsigned u = __builtin_bit_cast(unsigned, f);
  u += 0x7FFFu + ((u >> 16) & 1u);          // round-to-nearest-even
  return (unsigned short)(u >> 16);
}

static __device__ __forceinline__ v8f wmma_bf16(const Frag& a, const Frag& b, v8f c) {
  return __builtin_amdgcn_wmma_f32_16x16x32_bf16(
      false, a.v, false, b.v, (short)0, c, false, false);
}

// A-layout fragment: two 16B chunks at (k0 + half*16) and (k0 + half*16 + 8).
static __device__ __forceinline__ void load_fragA(Frag& f, const unsigned short* row,
                                                  int k0, int half) {
  f.q[0] = *(const uint4*)(row + k0 + half * 16);
  f.q[1] = *(const uint4*)(row + k0 + half * 16 + 8);
}
// B-layout fragment from K-major storage: one contiguous 32B run.
static __device__ __forceinline__ void load_fragB(Frag& f, const unsigned short* p) {
  const uint4* q = (const uint4*)p;
  f.q[0] = q[0];
  f.q[1] = q[1];
}

#if HAVE_TDM
// Issue a TDM load of a (tile1 x tile0) bf16 tile (tile1=0 -> 1-D run) from
// global (row stride = stride0 elements) into LDS at lds_off (packed rows).
// D# packing per CDNA5 ISA ch.8: group0 = {flags, lds_addr, global_addr, type},
// group1 = {data_size, tensor dims, tile dims, strides}.
static __device__ __forceinline__ void tdm_load_bf16(
    const unsigned short* gsrc, unsigned lds_off,
    int tile0, int tile1, unsigned stride0) {
  unsigned long long ga = (unsigned long long)gsrc;
  u32x4 g0;
  g0.x = 0x1u;                                       // count=1, user descriptor
  g0.y = lds_off;                                    // lds_addr (bytes)
  g0.z = (unsigned)ga;                               // global_addr[31:0]
  g0.w = (unsigned)((ga >> 32) & 0x01FFFFFFull) | (2u << 30);  // addr[56:32]|type=2
  i32x8 g1;
  g1[0] = (1 << 16);                                 // data_size=1 (2 bytes)
  g1[1] = 0;                                         // tensor_dim0 = 1<<20 (low16=0)
  g1[2] = 0x10 | (64 << 16);                         // tensor_dim0 hi | tensor_dim1=64
  g1[3] = (tile0 << 16);                             // tile_dim0
  g1[4] = tile1;                                     // tile_dim1 (0 -> unused/1D)
  g1[5] = (int)stride0;                              // tensor_dim0_stride[31:0]
  g1[6] = 0;
  g1[7] = 0;
  i32x4 z4 = {0, 0, 0, 0};
#if __clang_major__ >= 23
  i32x8 z8 = {0, 0, 0, 0, 0, 0, 0, 0};
  __builtin_amdgcn_tensor_load_to_lds(g0, g1, z4, z4, z8, 0);
#else
  __builtin_amdgcn_tensor_load_to_lds(g0, g1, z4, z4, 0);
#endif
}
#endif

// ---------------- fp32 -> bf16 conversion ----------------
__global__ void k_cvt(const float* __restrict__ s, unsigned short* __restrict__ d, int n) {
  int i = blockIdx.x * blockDim.x + threadIdx.x;
  if (i < n) d[i] = f2bf(s[i]);
}

// fp32 [K,N] -> bf16 transposed [N,K], LDS-tiled for coalescing
__global__ __launch_bounds__(1024) void k_cvtT(const float* __restrict__ src,
                                               unsigned short* __restrict__ dst,
                                               int K, int N) {
  __shared__ unsigned short tile[32][33];
  const int tx = threadIdx.x, ty = threadIdx.y;
  const int n0 = blockIdx.x * 32, k0 = blockIdx.y * 32;
  tile[ty][tx] = f2bf(src[(k0 + ty) * N + (n0 + tx)]);
  __syncthreads();
  dst[(n0 + ty) * K + (k0 + tx)] = tile[tx][ty];
}

// ---------------- QKV projection GEMM --------------------
// Xb [4096,1024] bf16, WqkvT [3072,1024] bf16 (K-major).
// Writes Q (pre-scaled 1/32) and K in [B*H,T,64]; V transposed in [B*H,64,T].
__global__ __launch_bounds__(128) void k_gemm_qkv(
    const unsigned short* __restrict__ Xb,
    const unsigned short* __restrict__ WqkvT,
    unsigned short* __restrict__ Qb,
    unsigned short* __restrict__ Kb,
    unsigned short* __restrict__ Vt) {
  const int lane = threadIdx.x & 31;
  const int half = lane >> 4;
  const int l16  = lane & 15;
  const int m0 = blockIdx.x * 32;
  const int n0 = (blockIdx.y * 4 + threadIdx.y) * 64;

  const unsigned short* Arow0 = Xb + (m0 + l16) * D_MODEL;
  const unsigned short* Arow1 = Xb + (m0 + 16 + l16) * D_MODEL;

  v8f acc[2][4] = {};
  for (int k0 = 0; k0 < D_MODEL; k0 += 32) {
    if (k0 + 32 < D_MODEL) __builtin_prefetch(Arow0 + k0 + 32, 0, 1);
    Frag a0, a1;
    load_fragA(a0, Arow0, k0, half);
    load_fragA(a1, Arow1, k0, half);
#pragma unroll
    for (int nt = 0; nt < 4; ++nt) {
      Frag b;
      load_fragB(b, WqkvT + (size_t)(n0 + nt * 16 + l16) * D_MODEL + k0 + half * 16);
      acc[0][nt] = wmma_bf16(a0, b, acc[0][nt]);
      acc[1][nt] = wmma_bf16(a1, b, acc[1][nt]);
    }
  }

#pragma unroll
  for (int nt = 0; nt < 4; ++nt) {
    const int colB = n0 + nt * 16 + l16;
    const int mat = colB >> 10;        // 0=Q 1=K 2=V (uniform per nt-tile)
    const int cc  = colB & 1023;
    const int h   = cc >> 6;
    const int d   = cc & 63;
#pragma unroll
    for (int rt = 0; rt < 2; ++rt) {
      if (mat == 2) {
        // V transposed: lane owns column d; 8 consecutive t -> one b128 store
        const int m  = m0 + rt * 16 + half * 8;
        const int bb = m >> 11;
        const int t  = m & (SEQ - 1);
        const int bh = bb * N_HEADS + h;
        uint4 pk;
        pk.x = (unsigned)f2bf(acc[rt][nt][0]) | ((unsigned)f2bf(acc[rt][nt][1]) << 16);
        pk.y = (unsigned)f2bf(acc[rt][nt][2]) | ((unsigned)f2bf(acc[rt][nt][3]) << 16);
        pk.z = (unsigned)f2bf(acc[rt][nt][4]) | ((unsigned)f2bf(acc[rt][nt][5]) << 16);
        pk.w = (unsigned)f2bf(acc[rt][nt][6]) | ((unsigned)f2bf(acc[rt][nt][7]) << 16);
        *(uint4*)(Vt + ((size_t)bh * HEAD_DIM + d) * SEQ + t) = pk;
      } else {
        unsigned short* dst = (mat == 0) ? Qb : Kb;
        const float scl = (mat == 0) ? 0.03125f : 1.0f;  // d_model^-0.5 folded into Q
#pragma unroll
        for (int r = 0; r < 8; ++r) {
          const int m  = m0 + rt * 16 + r + half * 8;
          const int bb = m >> 11;
          const int t  = m & (SEQ - 1);
          dst[(((bb * N_HEADS + h) * SEQ) + t) * HEAD_DIM + d] = f2bf(acc[rt][nt][r] * scl);
        }
      }
    }
  }
}

// ---------------- flash attention ------------------------
// One wave per block (so TDM descriptors are scalar); 16-row query tile,
// keys streamed 32 at a time through double-buffered TDM tiles in LDS.
__global__ __launch_bounds__(32) void k_attn(
    const unsigned short* __restrict__ Qb,   // [B*H,T,64]
    const unsigned short* __restrict__ Kb,   // [B*H,T,64]
    const unsigned short* __restrict__ Vtg,  // [B*H,64,T]
    unsigned short* __restrict__ Ob) {       // [4096,1024]
  __shared__ alignas(16) unsigned short Pl[16 * 32];
#if HAVE_TDM
  __shared__ alignas(16) unsigned short Ktile[2][32 * HEAD_DIM];  // 2 x 4KB
  __shared__ alignas(16) unsigned short Vtile[2][HEAD_DIM * 32];  // 2 x 4KB
#endif
  const int lane = threadIdx.x & 31;
  const int half = lane >> 4;
  const int l16  = lane & 15;
  const int i0 = blockIdx.x * 16;
  const int bh = blockIdx.y;                 // scalar (SGPR)
  const int b  = bh >> 4;
  const int h  = bh & (N_HEADS - 1);
  const unsigned short* Q = Qb  + (size_t)bh * SEQ * HEAD_DIM;
  const unsigned short* K = Kb  + (size_t)bh * SEQ * HEAD_DIM;
  const unsigned short* V = Vtg + (size_t)bh * HEAD_DIM * SEQ;

  Frag aQ0, aQ1;
  load_fragA(aQ0, Q + (i0 + l16) * HEAD_DIM, 0, half);
  load_fragA(aQ1, Q + (i0 + l16) * HEAD_DIM, 32, half);

  v8f o[4] = {};
  float mrow[8], lrow[8];
#pragma unroll
  for (int r = 0; r < 8; ++r) { mrow[r] = -3.0e38f; lrow[r] = 0.0f; }

  const int jEnd = i0 + 16;                 // causal: keys j < jEnd

#if HAVE_TDM
  // prime first K/V tiles: K tile = contiguous 32x64 (1-D 4KB run);
  // V^T tile = 2-D: 64 rows of 32 keys, row stride SEQ.
  tdm_load_bf16(K, (unsigned)(unsigned long long)&Ktile[0][0], 2048, 0, 2048);
  tdm_load_bf16(V, (unsigned)(unsigned long long)&Vtile[0][0], 32, 64, SEQ);
  __builtin_amdgcn_s_wait_tensorcnt(0);
#endif

  for (int js = 0; js < jEnd; js += 32) {
    const int idx = (js >> 5) & 1;
    const bool more = (js + 32) < jEnd;     // uniform
#if HAVE_TDM
    const unsigned short* Kt = &Ktile[idx][0];        // rows: key, stride 64
    const unsigned short* Vw = &Vtile[idx][0];        // rows: d,   stride 32
    const int vrs = 32, voff = 0;
    if (more) {                             // async prefetch next tiles
      tdm_load_bf16(K + (js + 32) * HEAD_DIM,
                    (unsigned)(unsigned long long)&Ktile[idx ^ 1][0], 2048, 0, 2048);
      tdm_load_bf16(V + (js + 32),
                    (unsigned)(unsigned long long)&Vtile[idx ^ 1][0], 32, 64, SEQ);
    }
#else
    const unsigned short* Kt = K + js * HEAD_DIM;
    const unsigned short* Vw = V + js;
    const int vrs = SEQ, voff = 0;
#endif

    v8f s0 = {}, s1 = {};
    {
      Frag bK;
      load_fragB(bK, Kt + l16 * HEAD_DIM + half * 16);
      s0 = wmma_bf16(aQ0, bK, s0);
      load_fragB(bK, Kt + l16 * HEAD_DIM + 32 + half * 16);
      s0 = wmma_bf16(aQ1, bK, s0);
      if (js + 16 < jEnd) {                 // uniform branch: EXEC stays full
        load_fragB(bK, Kt + (16 + l16) * HEAD_DIM + half * 16);
        s1 = wmma_bf16(aQ0, bK, s1);
        load_fragB(bK, Kt + (16 + l16) * HEAD_DIM + 32 + half * 16);
        s1 = wmma_bf16(aQ1, bK, s1);
      }
    }

    // causal mask + online softmax (row r: VGPR r across 16 lanes of one half)
#pragma unroll
    for (int r = 0; r < 8; ++r) {
      const int i = i0 + r + half * 8;
      if (js + l16 > i)      s0[r] = -3.0e38f;
      if (js + 16 + l16 > i) s1[r] = -3.0e38f;   // also masks never-computed tile 1
      float mx = fmaxf(s0[r], s1[r]);
#pragma unroll
      for (int off = 8; off >= 1; off >>= 1)
        mx = fmaxf(mx, __shfl_xor(mx, off, 32));
      const float mnew = fmaxf(mrow[r], mx);
      const float p0 = __expf(s0[r] - mnew);
      const float p1 = __expf(s1[r] - mnew);
      s0[r] = p0; s1[r] = p1;
      float rs = p0 + p1;
#pragma unroll
      for (int off = 8; off >= 1; off >>= 1)
        rs += __shfl_xor(rs, off, 32);
      const float alpha = __expf(mrow[r] - mnew);
      lrow[r] = lrow[r] * alpha + rs;
      mrow[r] = mnew;
      o[0][r] *= alpha; o[1][r] *= alpha; o[2][r] *= alpha; o[3][r] *= alpha;
    }

    // repack P: C/D layout -> A layout via LDS tile
    __syncthreads();
#pragma unroll
    for (int r = 0; r < 8; ++r) {
      Pl[(r + half * 8) * 32 + l16]      = f2bf(s0[r]);
      Pl[(r + half * 8) * 32 + 16 + l16] = f2bf(s1[r]);
    }
    __syncthreads();
    Frag aP;
    load_fragA(aP, &Pl[l16 * 32], 0, half);   // ds_load_b128 x2

    // O += P(16x32) * V(32x64): V^T rows are K-major -> b128 fragment loads
#pragma unroll
    for (int nt = 0; nt < 4; ++nt) {
      Frag bV;
      load_fragB(bV, Vw + (size_t)(nt * 16 + l16) * vrs + voff + half * 16);
      o[nt] = wmma_bf16(aP, bV, o[nt]);
    }

#if HAVE_TDM
    if (more) __builtin_amdgcn_s_wait_tensorcnt(0);  // close prefetch pipeline
#endif
  }

  // finalize: divide by l, write O [row=b*T+t, col=h*64+d] bf16
#pragma unroll
  for (int r = 0; r < 8; ++r) {
    const float inv = 1.0f / lrow[r];
    const int t = i0 + r + half * 8;
    unsigned short* dst = Ob + ((size_t)b * SEQ + t) * D_MODEL + h * HEAD_DIM;
    dst[ 0 + l16] = f2bf(o[0][r] * inv);
    dst[16 + l16] = f2bf(o[1][r] * inv);
    dst[32 + l16] = f2bf(o[2][r] * inv);
    dst[48 + l16] = f2bf(o[3][r] * inv);
  }
}

// ---------------- output projection GEMM -----------------
// Ab [4096,1024] bf16, WoT [1024,1024] bf16 (K-major) -> fp32 out
__global__ __launch_bounds__(128) void k_gemm_out(
    const unsigned short* __restrict__ Ab,
    const unsigned short* __restrict__ WoT,
    float* __restrict__ Out) {
  const int lane = threadIdx.x & 31;
  const int half = lane >> 4;
  const int l16  = lane & 15;
  const int m0 = blockIdx.x * 32;
  const int n0 = (blockIdx.y * 4 + threadIdx.y) * 64;

  const unsigned short* Arow0 = Ab + (m0 + l16) * D_MODEL;
  const unsigned short* Arow1 = Ab + (m0 + 16 + l16) * D_MODEL;

  v8f acc[2][4] = {};
  for (int k0 = 0; k0 < D_MODEL; k0 += 32) {
    if (k0 + 32 < D_MODEL) __builtin_prefetch(Arow0 + k0 + 32, 0, 1);
    Frag a0, a1;
    load_fragA(a0, Arow0, k0, half);
    load_fragA(a1, Arow1, k0, half);
#pragma unroll
    for (int nt = 0; nt < 4; ++nt) {
      Frag b;
      load_fragB(b, WoT + (size_t)(n0 + nt * 16 + l16) * D_MODEL + k0 + half * 16);
      acc[0][nt] = wmma_bf16(a0, b, acc[0][nt]);
      acc[1][nt] = wmma_bf16(a1, b, acc[1][nt]);
    }
  }
#pragma unroll
  for (int nt = 0; nt < 4; ++nt) {
    const int colB = n0 + nt * 16 + l16;
#pragma unroll
    for (int rt = 0; rt < 2; ++rt)
#pragma unroll
      for (int r = 0; r < 8; ++r) {
        const int m = m0 + rt * 16 + r + half * 8;
        Out[(size_t)m * D_MODEL + colB] = acc[rt][nt][r];
      }
  }
}

extern "C" void kernel_launch(void* const* d_in, const int* in_sizes, int n_in,
                              void* d_out, int out_size, void* d_ws, size_t ws_size,
                              hipStream_t stream) {
  const float* x    = (const float*)d_in[0];   // [2,2048,1024]
  const float* Wqkv = (const float*)d_in[1];   // [1024,3072]
  const float* Wo   = (const float*)d_in[2];   // [1024,1024]
  float* out = (float*)d_out;

  // workspace (bf16 as ushort): 24M elements = 48 MB
  unsigned short* ws    = (unsigned short*)d_ws;
  unsigned short* Xb    = ws;                                   // 4096*1024
  unsigned short* WqkvT = Xb    + (size_t)M_TOTAL * D_MODEL;    // [3072,1024]
  unsigned short* WoT   = WqkvT + (size_t)D_MODEL * N_QKV;      // [1024,1024]
  unsigned short* Qb    = WoT   + (size_t)D_MODEL * D_MODEL;    // [B*H,T,64]
  unsigned short* Kb    = Qb    + (size_t)M_TOTAL * D_MODEL;    // [B*H,T,64]
  unsigned short* Vt    = Kb    + (size_t)M_TOTAL * D_MODEL;    // [B*H,64,T]
  unsigned short* Ob    = Vt    + (size_t)M_TOTAL * D_MODEL;    // [4096,1024]

  // 1) conversions (+ weight transposes to K-major)
  {
    int n = M_TOTAL * D_MODEL;
    k_cvt<<<dim3((n + 255) / 256), dim3(256), 0, stream>>>(x, Xb, n);
    k_cvtT<<<dim3(N_QKV / 32, D_MODEL / 32), dim3(32, 32), 0, stream>>>(
        Wqkv, WqkvT, D_MODEL, N_QKV);
    k_cvtT<<<dim3(D_MODEL / 32, D_MODEL / 32), dim3(32, 32), 0, stream>>>(
        Wo, WoT, D_MODEL, D_MODEL);
  }

  // 2) QKV projection: 32x64 tile per wave
  k_gemm_qkv<<<dim3(M_TOTAL / 32, N_QKV / 64 / 4), dim3(32, 4), 0, stream>>>(
      Xb, WqkvT, Qb, Kb, Vt);

  // 3) flash attention (1 wave/block; TDM double-buffered K/V tiles)
  k_attn<<<dim3(SEQ / 16, BATCH * N_HEADS), dim3(32), 0, stream>>>(
      Qb, Kb, Vt, Ob);

  // 4) output projection -> fp32
  k_gemm_out<<<dim3(M_TOTAL / 32, D_MODEL / 64 / 4), dim3(32, 4), 0, stream>>>(
      Ob, WoT, out);
}
